// GlobalSuperNodeLayer_12678743458336
// MI455X (gfx1250) — matrix operand order repaired and checked by reading, not compiled, not run on hardware
//
#include <hip/hip_runtime.h>

typedef __attribute__((ext_vector_type(16))) _Float16 v16h;
typedef __attribute__((ext_vector_type(8)))  float    v8f;
typedef __attribute__((ext_vector_type(4)))  float    v4f;

// ---- configuration -------------------------------------------------------
#define BUS_BLOCKS  128   // 1024 waves
#define GEN_BLOCKS   32   //  256 waves
#define LOAD_BLOCKS  64   //  512 waves
#define SLOTS_BUS  (BUS_BLOCKS * 8)
#define SLOTS_GEN  (GEN_BLOCKS * 8)
#define SLOTS_LOAD (LOAD_BLOCKS * 8)
#define SLOT_F 68   // floats per pool partial slot: xbar[64], m, s, pad -> 272B (16B aligned)

// ---- fragment loaders ----------------------------------------------------
// A fragment (16x32 f16, M x K): lane l%16 = row; lanes<16 hold K 0..15,
// lanes>=16 hold K 16..31 (relative to the 32-wide k-step).
__device__ __forceinline__ v16h load_a_frag(const float* __restrict__ xrow,
                                            int ks, int ka) {
  const v4f* p = (const v4f*)(xrow + ks * 32 + ka);
  v4f a0 = __builtin_nontemporal_load(p + 0);
  v4f a1 = __builtin_nontemporal_load(p + 1);
  v4f a2 = __builtin_nontemporal_load(p + 2);
  v4f a3 = __builtin_nontemporal_load(p + 3);
  v16h r;
#pragma unroll
  for (int i = 0; i < 4; ++i) {
    r[i]      = (_Float16)a0[i];
    r[4 + i]  = (_Float16)a1[i];
    r[8 + i]  = (_Float16)a2[i];
    r[12 + i] = (_Float16)a3[i];
  }
  return r;
}

// B fragment (32x16 f16, K x N) from transposed f16 weights WT[n][k] (k=0..63):
// lane l%16 = column; lanes<16 hold K 0..15, lanes>=16 hold K 16..31.
__device__ __forceinline__ v16h load_b_frag(const _Float16* __restrict__ WT,
                                            int t, int ks, int lane) {
  int col = t * 16 + (lane & 15);
  int kb  = ks * 32 + ((lane < 16) ? 0 : 16);
  return *(const v16h*)(WT + col * 64 + kb);
}

// ---- kernel 0: prep ------------------------------------------------------
// (a) WnT[type][n][k] = f16(Wn[type][k][n]) for k<64 (transposed K-slice)
// (b) q[type][k] = sum_n Wp[type][k][n] * g[n]   (folded logit vector)
// (c) c0[type]   = sum_n bp[type][n] * g[n]      (folded logit bias)
__global__ __launch_bounds__(256) void gsn_prep(
    const float* __restrict__ Wp0, const float* __restrict__ Wp1,
    const float* __restrict__ Wp2, const float* __restrict__ bp0,
    const float* __restrict__ bp1, const float* __restrict__ bp2,
    const float* __restrict__ g,
    const float* __restrict__ Wn0, const float* __restrict__ Wn1,
    const float* __restrict__ Wn2,
    _Float16* __restrict__ WnT, float* __restrict__ qbuf,
    float* __restrict__ c0buf) {
  const float* Wp[3] = {Wp0, Wp1, Wp2};
  const float* bp[3] = {bp0, bp1, bp2};
  const float* Wn[3] = {Wn0, Wn1, Wn2};
  const int tid0   = blockIdx.x * blockDim.x + threadIdx.x;
  const int stride = blockDim.x * gridDim.x;
  for (int i = tid0; i < 128 * 64; i += stride) {
    int n = i >> 6, k = i & 63;
#pragma unroll
    for (int t = 0; t < 3; ++t)
      WnT[t * 8192 + i] = (_Float16)Wn[t][k * 128 + n];  // rows 0..63 of Wn
  }
  for (int i = tid0; i < 3 * 64; i += stride) {
    int t = i >> 6, k = i & 63;
    float a = 0.f;
    for (int n = 0; n < 128; ++n) a += Wp[t][k * 128 + n] * g[n];
    qbuf[i] = a;
  }
  for (int i = tid0; i < 3; i += stride) {
    float a = 0.f;
    for (int n = 0; n < 128; ++n) a += bp[i][n] * g[n];
    c0buf[i] = a;
  }
}

// ---- kernel 1: streaming online-softmax pooling over x -------------------
// Identity used: logit_i = x_i . q + c0 ;  pooled = (sum_i alpha_i x_i) @ Wp + bp.
// One row per lane, per-lane online softmax; wave-level merge at the end.
__global__ __launch_bounds__(256) void gsn_pool(
    const float* __restrict__ x, int nrows,
    const float* __restrict__ q, const float* __restrict__ c0p,
    float* __restrict__ partials) {
  const int lane    = threadIdx.x & 31;
  const int gthread = blockIdx.x * blockDim.x + threadIdx.x;
  const int wave    = gthread >> 5;
  const int stride  = gridDim.x * blockDim.x;

  v4f Q[16];
#pragma unroll
  for (int i = 0; i < 16; ++i) Q[i] = ((const v4f*)q)[i];
  const float c0 = c0p[0];

  float m = -1e30f, s = 0.f;
  v4f acc[16];
#pragma unroll
  for (int i = 0; i < 16; ++i) acc[i] = (v4f){0.f, 0.f, 0.f, 0.f};

  const v4f* xr = (const v4f*)(x + (size_t)gthread * 64);
  const size_t xstep = (size_t)stride * 16;   // in v4f units
  for (int row = gthread; row < nrows; row += stride, xr += xstep) {
    v4f rx[16];
#pragma unroll
    for (int i = 0; i < 16; ++i) rx[i] = __builtin_nontemporal_load(xr + i);

    v4f d = (v4f){0.f, 0.f, 0.f, 0.f};
#pragma unroll
    for (int i = 0; i < 16; ++i) d += rx[i] * Q[i];
    float dot = c0 + d[0] + d[1] + d[2] + d[3];

    if (dot > m) {                 // rescale running state (rare after warmup)
      float sc = __expf(m - dot);
      s *= sc;
#pragma unroll
      for (int i = 0; i < 16; ++i) acc[i] *= sc;
      m = dot;
    }
    float w = __expf(dot - m);
    s += w;
#pragma unroll
    for (int i = 0; i < 16; ++i) acc[i] += rx[i] * w;
  }

  // ---- merge 32 lanes -> one wave partial ----
  float wm = m;
#pragma unroll
  for (int k = 1; k < 32; k <<= 1) wm = fmaxf(wm, __shfl_xor(wm, k, 32));
  float sc = __expf(m - wm);
  s *= sc;
#pragma unroll
  for (int i = 0; i < 16; ++i) acc[i] *= sc;
#pragma unroll
  for (int k = 1; k < 32; k <<= 1) s += __shfl_xor(s, k, 32);
#pragma unroll
  for (int i = 0; i < 16; ++i)
#pragma unroll
    for (int j = 0; j < 4; ++j) {
      float v = acc[i][j];
#pragma unroll
      for (int k = 1; k < 32; k <<= 1) v += __shfl_xor(v, k, 32);
      acc[i][j] = v;
    }

  float* slot = partials + (size_t)wave * SLOT_F;
  if (lane == 0) {
    v4f* sp = (v4f*)slot;
#pragma unroll
    for (int i = 0; i < 16; ++i) sp[i] = acc[i];
    slot[64] = wm;
    slot[65] = s;
  }
}

// ---- kernel 2: combine partials, pooled, new_g, h[type] ------------------
__global__ __launch_bounds__(256) void gsn_reduce(
    const float* __restrict__ pb, int nb,
    const float* __restrict__ pg, int ng,
    const float* __restrict__ pl, int nl,
    const float* __restrict__ g, const float* __restrict__ Wg,
    const float* __restrict__ bg,
    const float* __restrict__ Wp0, const float* __restrict__ bp0,
    const float* __restrict__ Wp1, const float* __restrict__ bp1,
    const float* __restrict__ Wp2, const float* __restrict__ bp2,
    const float* __restrict__ Wn0, const float* __restrict__ bn0,
    const float* __restrict__ Wn1, const float* __restrict__ bn1,
    const float* __restrict__ Wn2, const float* __restrict__ bn2,
    float* __restrict__ out_g, float* __restrict__ hbuf) {
  __shared__ float red[256];
  __shared__ float xbar[64];
  __shared__ float allp[128];
  __shared__ float ngv[128];
  const int tid = threadIdx.x;
  if (tid < 128) allp[tid] = 0.f;
  __syncthreads();

  const float* P3[3]  = {pb, pg, pl};
  const int    N3[3]  = {nb, ng, nl};
  const float* Wp3[3] = {Wp0, Wp1, Wp2};
  const float* bp3[3] = {bp0, bp1, bp2};

  for (int t = 0; t < 3; ++t) {
    const float* P = P3[t];
    const int n = N3[t];

    float m = -1e30f;
    for (int i = tid; i < n; i += 256) m = fmaxf(m, P[(size_t)i * SLOT_F + 64]);
    red[tid] = m; __syncthreads();
    for (int s2 = 128; s2 > 0; s2 >>= 1) {
      if (tid < s2) red[tid] = fmaxf(red[tid], red[tid + s2]);
      __syncthreads();
    }
    const float M = red[0]; __syncthreads();

    float s = 0.f;
    for (int i = tid; i < n; i += 256)
      s += __expf(P[(size_t)i * SLOT_F + 64] - M) * P[(size_t)i * SLOT_F + 65];
    red[tid] = s; __syncthreads();
    for (int s2 = 128; s2 > 0; s2 >>= 1) {
      if (tid < s2) red[tid] += red[tid + s2];
      __syncthreads();
    }
    const float S = red[0]; __syncthreads();

    if (tid < 64) {
      float a = 0.f;
      for (int i = 0; i < n; ++i)
        a += __expf(P[(size_t)i * SLOT_F + 64] - M) * P[(size_t)i * SLOT_F + tid];
      xbar[tid] = a / S;            // softmax-weighted mean of x rows
    }
    __syncthreads();
    if (tid < 128) {
      float a = bp3[t][tid];
      for (int k = 0; k < 64; ++k) a += xbar[k] * Wp3[t][k * 128 + tid];
      allp[tid] += a;               // pooled_t = xbar @ Wp_t + bp_t
    }
    __syncthreads();
  }

  if (tid < 128) {
    float a = bg[tid];
    for (int k = 0; k < 128; ++k) a += g[k] * Wg[k * 128 + tid];
    for (int k = 0; k < 128; ++k) a += allp[k] * Wg[(128 + k) * 128 + tid];
    a = fmaxf(a, 0.f);
    ngv[tid]   = a;
    out_g[tid] = a;
  }
  __syncthreads();

  const float* Wn3[3] = {Wn0, Wn1, Wn2};
  const float* bn3[3] = {bn0, bn1, bn2};
  for (int t = 0; t < 3; ++t) {
    if (tid < 128) {
      float a = bn3[t][tid];
      for (int k = 0; k < 128; ++k) a += ngv[k] * Wn3[t][(64 + k) * 128 + tid];
      hbuf[t * 128 + tid] = a;      // h_t = new_g @ Wn_t[64:] + bn_t
    }
  }
}

// ---- kernel 3: out = relu(x @ Wn[:64] + h) via WMMA ----------------------
// Stores use one per-lane base pointer per tile + constant immediate offsets.
__global__ __launch_bounds__(256) void gsn_update(
    const float* __restrict__ x, int nrows,
    const _Float16* __restrict__ WT, const float* __restrict__ h,
    float* __restrict__ out) {
  const int lane   = threadIdx.x & 31;
  const int wave   = (blockIdx.x * blockDim.x + threadIdx.x) >> 5;
  const int nwaves = (gridDim.x * blockDim.x) >> 5;

  v16h B[8][2];
#pragma unroll
  for (int t = 0; t < 8; ++t) {
    B[t][0] = load_b_frag(WT, t, 0, lane);
    B[t][1] = load_b_frag(WT, t, 1, lane);
  }
  float hv[8];
#pragma unroll
  for (int t = 0; t < 8; ++t) hv[t] = h[t * 16 + (lane & 15)];

  const int ntiles = nrows >> 4;               // row counts are multiples of 16
  const int ka = (lane < 16) ? 0 : 16;
  const int rh = (lane < 16) ? 0 : 8;

  const float* xr = x + (size_t)(wave * 16 + (lane & 15)) * 64;
  float*       ob = out + (size_t)(wave * 16 + rh) * 128 + (lane & 15);
  const size_t xstep = (size_t)nwaves * 16 * 64;
  const size_t ostep = (size_t)nwaves * 16 * 128;

  for (int tile = wave; tile < ntiles; tile += nwaves, xr += xstep, ob += ostep) {
    v16h A0 = load_a_frag(xr, 0, ka);
    v16h A1 = load_a_frag(xr, 1, ka);

#pragma unroll
    for (int t = 0; t < 8; ++t) {
      v8f c = {};
      c = __builtin_amdgcn_wmma_f32_16x16x32_f16(false, A0, false, B[t][0],
                                                 (short)0, c, false, false);
      c = __builtin_amdgcn_wmma_f32_16x16x32_f16(false, A1, false, B[t][1],
                                                 (short)0, c, false, false);
#pragma unroll
      for (int r = 0; r < 8; ++r)
        __builtin_nontemporal_store(fmaxf(c[r] + hv[t], 0.f),
                                    ob + (t * 16 + r * 128));  // const offsets
    }
  }
}

// ---- launcher ------------------------------------------------------------
extern "C" void kernel_launch(void* const* d_in, const int* in_sizes, int n_in,
                              void* d_out, int out_size, void* d_ws,
                              size_t ws_size, hipStream_t stream) {
  const float* x_bus  = (const float*)d_in[0];
  const float* x_gen  = (const float*)d_in[1];
  const float* x_load = (const float*)d_in[2];
  const float* g      = (const float*)d_in[3];
  const float* Wp_bus = (const float*)d_in[4];
  const float* bp_bus = (const float*)d_in[5];
  const float* Wp_gen = (const float*)d_in[6];
  const float* bp_gen = (const float*)d_in[7];
  const float* Wp_load= (const float*)d_in[8];
  const float* bp_load= (const float*)d_in[9];
  const float* Wg     = (const float*)d_in[10];
  const float* bg     = (const float*)d_in[11];
  const float* Wn_bus = (const float*)d_in[12];
  const float* bn_bus = (const float*)d_in[13];
  const float* Wn_gen = (const float*)d_in[14];
  const float* bn_gen = (const float*)d_in[15];
  const float* Wn_load= (const float*)d_in[16];
  const float* bn_load= (const float*)d_in[17];

  const int nb = in_sizes[0] / 64;
  const int ng = in_sizes[1] / 64;
  const int nl = in_sizes[2] / 64;

  // ---- workspace layout (all 16B-aligned) ----
  char* ws = (char*)d_ws;
  _Float16* WnT  = (_Float16*)ws;                    // 3*8192 f16   = 49152 B
  float*    qbuf = (float*)(ws + 49152);             // 3*64  f32    =   768 B
  float*    c0b  = (float*)(ws + 49920);             // 3 f32 (+pad) =    16 B
  float*    hbuf = (float*)(ws + 49936);             // 3*128 f32    =  1536 B
  float*    part_bus  = (float*)(ws + 51472);
  float*    part_gen  = part_bus + (size_t)SLOTS_BUS * SLOT_F;
  float*    part_load = part_gen + (size_t)SLOTS_GEN * SLOT_F;

  float* out      = (float*)d_out;
  float* out_g    = out;
  float* out_bus  = out + 128;
  float* out_gen  = out_bus + (size_t)nb * 128;
  float* out_load = out_gen + (size_t)ng * 128;

  gsn_prep<<<32, 256, 0, stream>>>(Wp_bus, Wp_gen, Wp_load,
                                   bp_bus, bp_gen, bp_load, g,
                                   Wn_bus, Wn_gen, Wn_load,
                                   WnT, qbuf, c0b);

  gsn_pool<<<BUS_BLOCKS, 256, 0, stream>>>(x_bus, nb, qbuf,       c0b,     part_bus);
  gsn_pool<<<GEN_BLOCKS, 256, 0, stream>>>(x_gen, ng, qbuf + 64,  c0b + 1, part_gen);
  gsn_pool<<<LOAD_BLOCKS,256, 0, stream>>>(x_load, nl, qbuf + 128, c0b + 2, part_load);

  gsn_reduce<<<1, 256, 0, stream>>>(part_bus, SLOTS_BUS, part_gen, SLOTS_GEN,
                                    part_load, SLOTS_LOAD, g, Wg, bg,
                                    Wp_bus, bp_bus, Wp_gen, bp_gen,
                                    Wp_load, bp_load,
                                    Wn_bus, bn_bus, Wn_gen, bn_gen,
                                    Wn_load, bn_load, out_g, hbuf);

  gsn_update<<<512, 256, 0, stream>>>(x_bus, nb, WnT,         hbuf,       out_bus);
  gsn_update<<<128, 256, 0, stream>>>(x_gen, ng, WnT + 8192,  hbuf + 128, out_gen);
  gsn_update<<<256, 256, 0, stream>>>(x_load, nl, WnT + 16384, hbuf + 256, out_load);
}
